// GATlayer_68728066671087
// MI455X (gfx1250) — matrix operand order repaired and checked by reading, not compiled, not run on hardware
//
#include <hip/hip_runtime.h>

// CDNA5 / gfx1250 — wave32, WMMA 16x16x32 bf16 with fp32 accumulate.
typedef __attribute__((ext_vector_type(16))) __bf16 v16bf;
typedef __attribute__((ext_vector_type(8)))  float  v8f;
typedef __attribute__((ext_vector_type(4)))  float  v4f;

#define LEAKY_ALPHA 0.2f

// -------------------------------------------------------------------------
// Kernel 0: W (f32, [Fin,Fout]) -> Wt (bf16, transposed [Fout,Fin]).
// -------------------------------------------------------------------------
__global__ void gat_cvt_w_kernel(const float* __restrict__ W,
                                 __bf16* __restrict__ Wt, int Fin, int Fout) {
    int i = blockIdx.x * blockDim.x + threadIdx.x;  // i = o*Fin + k
    if (i < Fin * Fout) {
        int o = i / Fin;
        int k = i - o * Fin;
        Wt[i] = (__bf16)W[k * Fout + o];
    }
}

// -------------------------------------------------------------------------
// Kernel 1: h = x @ W + b (bf16 WMMA, f32 accum); a = h.wa, c = h.wb.
// h written TRANSPOSED as h_t[b][feat][node] (bf16). One wave per tile.
// All 8 B-fragments of a K-chunk are preloaded before the WMMA chain so
// loads overlap with matrix ops via partial s_wait_loadcnt.
// -------------------------------------------------------------------------
__global__ __launch_bounds__(128) void gat_proj_kernel(
    const float* __restrict__ x, const __bf16* __restrict__ Wt,
    const float* __restrict__ b, const float* __restrict__ wa,
    const float* __restrict__ wb, __bf16* __restrict__ h_t,
    float* __restrict__ a_out, float* __restrict__ c_out,
    int B, int N, int Fin, int Fout) {
    const int lane  = threadIdx.x & 31;
    const int wave  = threadIdx.x >> 5;
    const int tile  = blockIdx.x * (blockDim.x >> 5) + wave;
    const int tilesPerBatch = N >> 4;
    const int bIdx  = tile / tilesPerBatch;
    const int r0    = (tile - bIdx * tilesPerBatch) << 4;
    const int lmod  = lane & 15;
    const int lhalf = lane >> 4;

    v8f acc[8];
#pragma unroll
    for (int f = 0; f < 8; ++f) {
        float bv = b[f * 16 + lmod];
#pragma unroll
        for (int v = 0; v < 8; ++v) acc[f][v] = bv;
    }

    const float* xrow = x + ((size_t)bIdx * N + r0 + lmod) * Fin;

    for (int kk = 0; kk < Fin; kk += 32) {
        // Preload all 8 B-fragments (one 16-load clause).
        v16bf bfr[8];
#pragma unroll
        for (int f = 0; f < 8; ++f)
            bfr[f] = *(const v16bf*)(Wt + (size_t)(f * 16 + lmod) * Fin +
                                     kk + 16 * lhalf);
        // A-fragment: two contiguous 8-float runs -> 4 float4 loads + pk cvt.
        v16bf afrag;
#pragma unroll
        for (int g = 0; g < 2; ++g) {
            const v4f* p = (const v4f*)(xrow + kk + 16 * lhalf + 8 * g);
            v4f x0 = p[0];
            v4f x1 = p[1];
#pragma unroll
            for (int j = 0; j < 4; ++j) {
                afrag[g * 8 + j]     = (__bf16)x0[j];
                afrag[g * 8 + 4 + j] = (__bf16)x1[j];
            }
        }
#pragma unroll
        for (int f = 0; f < 8; ++f)
            acc[f] = __builtin_amdgcn_wmma_f32_16x16x32_bf16(
                false, afrag, false, bfr[f], (short)0, acc[f], false, false);
    }

    // Store h transposed (bf16); reduce a = h.wa, c = h.wb.
    float pa[8], pc[8];
#pragma unroll
    for (int v = 0; v < 8; ++v) { pa[v] = 0.f; pc[v] = 0.f; }

#pragma unroll
    for (int f = 0; f < 8; ++f) {
        float wav = wa[f * 16 + lmod];
        float wbv = wb[f * 16 + lmod];
        __bf16* hcol = h_t + ((size_t)bIdx * Fout + f * 16 + lmod) * N + r0;
#pragma unroll
        for (int v = 0; v < 8; ++v) {
            float hv = acc[f][v];                 // C/D: row = v + 8*lhalf
            hcol[v + 8 * lhalf] = (__bf16)hv;
            pa[v] += hv * wav;
            pc[v] += hv * wbv;
        }
    }
#pragma unroll
    for (int m = 1; m <= 8; m <<= 1) {            // masks 1..8 stay in-half
#pragma unroll
        for (int v = 0; v < 8; ++v) {
            pa[v] += __shfl_xor(pa[v], m, 32);
            pc[v] += __shfl_xor(pc[v], m, 32);
        }
    }
    if (lmod == 0) {
#pragma unroll
        for (int v = 0; v < 8; ++v) {
            int row = r0 + v + 8 * lhalf;
            a_out[(size_t)bIdx * N + row] = pa[v];
            c_out[(size_t)bIdx * N + row] = pc[v];
        }
    }
}

// -------------------------------------------------------------------------
// Kernel 2: fused masked-leaky-softmax + attn @ h (bf16 WMMA) + ReLU.
// Phase 1: pure fma/max row-max sweep (no exp).
// Phase 2: exp once per element -> bf16 A-fragment AND f32 row sum;
//          out = relu(acc / rowsum) at store (normalization is linear).
// 8 B-fragments per chunk preloaded before the WMMA chain.
// -------------------------------------------------------------------------
__global__ __launch_bounds__(128) void gat_attn_kernel(
    const float* __restrict__ mask, const float* __restrict__ a_in,
    const float* __restrict__ c_in, const __bf16* __restrict__ h_t,
    const float* __restrict__ ab, float* __restrict__ out,
    int B, int N, int Fout) {
    const int lane  = threadIdx.x & 31;
    const int wave  = threadIdx.x >> 5;
    const int tile  = blockIdx.x * (blockDim.x >> 5) + wave;
    const int tilesPerBatch = N >> 4;
    const int bIdx  = tile / tilesPerBatch;
    const int r0    = (tile - bIdx * tilesPerBatch) << 4;
    const int lmod  = lane & 15;
    const int lhalf = lane >> 4;

    const float abias = ab[0];
    const float* arow = a_in + (size_t)bIdx * N;

    float crow[16];
#pragma unroll
    for (int r = 0; r < 16; ++r) crow[r] = c_in[(size_t)bIdx * N + r0 + r];

    // ---- Phase 1: row max over all N columns (no exp, float4 loads) ----
    float mrow[16];
#pragma unroll
    for (int r = 0; r < 16; ++r) mrow[r] = -3.0e38f;

    for (int c4 = lane * 4; c4 < N; c4 += 128) {
        __builtin_prefetch(mask + (size_t)r0 * N + c4 + 1024, 0, 1);
        v4f av = *(const v4f*)(arow + c4);
#pragma unroll
        for (int r = 0; r < 16; ++r) {
            v4f mv = *(const v4f*)(mask + (size_t)(r0 + r) * N + c4);
#pragma unroll
            for (int j = 0; j < 4; ++j) {
                float ev = (av[j] + crow[r] + abias) * mv[j];
                ev = ev >= 0.f ? ev : LEAKY_ALPHA * ev;
                mrow[r] = fmaxf(mrow[r], ev);
            }
        }
    }
#pragma unroll
    for (int off = 1; off < 32; off <<= 1)
#pragma unroll
        for (int r = 0; r < 16; ++r)
            mrow[r] = fmaxf(mrow[r], __shfl_xor(mrow[r], off, 32));

    // This lane's A-fragment row is lmod: select its scalars.
    float msel = mrow[0], csel = crow[0];
#pragma unroll
    for (int r = 1; r < 16; ++r)
        if (lmod == r) { msel = mrow[r]; csel = crow[r]; }

    // ---- Phase 2: unnormalized attn @ h via bf16 WMMA + f32 row sums ----
    v8f oacc[8];
#pragma unroll
    for (int f = 0; f < 8; ++f)
#pragma unroll
        for (int v = 0; v < 8; ++v) oacc[f][v] = 0.f;

    float psum = 0.f;
    const float* maskA = mask + (size_t)(r0 + lmod) * N;
    const __bf16* hbase = h_t + (size_t)bIdx * Fout * N;

    for (int kc = 0; kc < N; kc += 32) {
        __builtin_prefetch(maskA + kc + 512, 0, 1);
        // Preload all 8 B-fragments (16-load clause; overlaps with WMMAs).
        v16bf bfr[8];
#pragma unroll
        for (int f = 0; f < 8; ++f)
            bfr[f] = *(const v16bf*)(hbase + (size_t)(f * 16 + lmod) * N +
                                     kc + 16 * lhalf);
        // Build attn A-fragment in f32, accumulate row sum.
        v16bf afrag;
#pragma unroll
        for (int g = 0; g < 2; ++g) {
            int cb = kc + 16 * lhalf + 8 * g;        // A 16x32: k = 16*lhalf + s
            v4f m0 = *(const v4f*)(maskA + cb);
            v4f m1 = *(const v4f*)(maskA + cb + 4);
            v4f a0 = *(const v4f*)(arow + cb);
            v4f a1 = *(const v4f*)(arow + cb + 4);
#pragma unroll
            for (int j = 0; j < 4; ++j) {
                float e0 = (a0[j] + csel + abias) * m0[j];
                e0 = e0 >= 0.f ? e0 : LEAKY_ALPHA * e0;
                float p0 = __expf(e0 - msel);
                psum += p0;
                afrag[g * 8 + j] = (__bf16)p0;

                float e1 = (a1[j] + csel + abias) * m1[j];
                e1 = e1 >= 0.f ? e1 : LEAKY_ALPHA * e1;
                float p1 = __expf(e1 - msel);
                psum += p1;
                afrag[g * 8 + 4 + j] = (__bf16)p1;
            }
        }
#pragma unroll
        for (int f = 0; f < 8; ++f)
            oacc[f] = __builtin_amdgcn_wmma_f32_16x16x32_bf16(
                false, afrag, false, bfr[f], (short)0, oacc[f], false, false);
    }

    // Row sums: lanes l and l^16 cover the same row (lmod).
    psum += __shfl_xor(psum, 16, 32);
    float sinv[16];
#pragma unroll
    for (int r = 0; r < 16; ++r) sinv[r] = 1.0f / __shfl(psum, r, 32);

    // Normalize + ReLU + store (C/D layout: row = v + 8*lhalf).
#pragma unroll
    for (int f = 0; f < 8; ++f)
#pragma unroll
        for (int v = 0; v < 8; ++v) {
            float is  = lhalf ? sinv[v + 8] : sinv[v];
            int   row = r0 + v + 8 * lhalf;
            out[((size_t)bIdx * N + row) * Fout + f * 16 + lmod] =
                fmaxf(oacc[f][v] * is, 0.f);
        }
}

// -------------------------------------------------------------------------
extern "C" void kernel_launch(void* const* d_in, const int* in_sizes, int n_in,
                              void* d_out, int out_size, void* d_ws, size_t ws_size,
                              hipStream_t stream) {
    const float* x    = (const float*)d_in[0];
    const float* mask = (const float*)d_in[1];
    const float* W    = (const float*)d_in[2];
    const float* b    = (const float*)d_in[3];
    const float* wa   = (const float*)d_in[4];
    const float* wb   = (const float*)d_in[5];
    const float* ab   = (const float*)d_in[6];
    float* out = (float*)d_out;

    const int B = 4, N = 4096, Fin = 256, Fout = 128;

    // Workspace layout: Wt (bf16, transposed) | h_t (bf16, transposed) | a | c
    char*   ws   = (char*)d_ws;
    __bf16* Wt   = (__bf16*)ws;
    size_t  off  = (size_t)Fin * Fout * sizeof(__bf16);          // 64 KB
    __bf16* h_t  = (__bf16*)(ws + off);
    off += (size_t)B * N * Fout * sizeof(__bf16);                // 4 MB
    float*  a_ws = (float*)(ws + off);
    off += (size_t)B * N * sizeof(float);
    float*  c_ws = (float*)(ws + off);

    gat_cvt_w_kernel<<<(Fin * Fout + 255) / 256, 256, 0, stream>>>(
        W, Wt, Fin, Fout);

    const int tiles = B * (N / 16);           // 1024 wave-tiles
    dim3 blk(128);                            // 4 waves per block
    gat_proj_kernel<<<tiles / 4, blk, 0, stream>>>(
        x, Wt, b, wa, wb, h_t, a_ws, c_ws, B, N, Fin, Fout);
    gat_attn_kernel<<<tiles / 4, blk, 0, stream>>>(
        mask, a_ws, c_ws, h_t, ab, out, B, N, Fout);
}
